// PonitaGen_27908697489434
// MI455X (gfx1250) — matrix operand order repaired and checked by reading, not compiled, not run on hardware
//
#include <hip/hip_runtime.h>

#define BB   2
#define NN   768
#define NIN  16
#define HID  64
#define BAS  32
#define WHID 256
#define POLY 14

typedef __attribute__((ext_vector_type(16))) __bf16 bf16x16;
typedef __attribute__((ext_vector_type(8)))  float  f32x8;

union Frag { bf16x16 v; uint4 q[2]; __bf16 h[16]; };

__device__ __forceinline__ f32x8 wmma_bf16(bf16x16 a, bf16x16 b, f32x8 c) {
  return __builtin_amdgcn_wmma_f32_16x16x32_bf16(false, a, false, b, (short)0, c, false, false);
}
__device__ __forceinline__ f32x8 zero8() {
  f32x8 z;
#pragma unroll
  for (int i = 0; i < 8; ++i) z[i] = 0.0f;
  return z;
}
// jax.nn.gelu default (approximate=True, tanh form), branch-free:
// tanh(u) = 1 - 2/(1+exp(2u)); exp via v_exp_f32 (exp2), rcp via v_rcp_f32.
// Limits: u>>0 -> exp2 = +inf -> rcp = 0 -> tanh = 1; u<<0 -> exp2 = 0 -> tanh = -1.
__device__ __forceinline__ float gelu_t(float x) {
  float u = 0.7978845608028654f * (x + 0.044715f * x * x * x);
  float e = __builtin_amdgcn_exp2f(u * 2.8853900817779268f); // 2*log2(e)
  float t = 1.0f - 2.0f * __builtin_amdgcn_rcpf(1.0f + e);
  return 0.5f * x * (1.0f + t);
}
#define WAIT_DS() asm volatile("s_wait_dscnt 0" ::: "memory")

// ---------------------------------------------------------------------------
// Prep: stem (h0 = a @ stem_w, stored transposed bf16 [b][c][s]) + weight
// swizzles into WMMA fragment order (lane-major, contiguous 32B per lane).
// B-frag layout (bf16 K32xN16): lane L: n=L&15, kbase=(L<16)?0:8, h[j]=W[kbase+j+(j&8)][n]
// ---------------------------------------------------------------------------
__global__ void __launch_bounds__(256) prep_kernel(
    const float* __restrict__ a, const float* __restrict__ stem_w,
    const float* __restrict__ kb_w1, const float* __restrict__ kb_w2,
    const float* __restrict__ w1_0, const float* __restrict__ w2_0, const float* __restrict__ cw_0,
    const float* __restrict__ w1_1, const float* __restrict__ w2_1, const float* __restrict__ cw_1,
    __bf16* __restrict__ ht0, __bf16* __restrict__ kbw1f, __bf16* __restrict__ kbw2f,
    __bf16* __restrict__ w1f0, __bf16* __restrict__ w2f0, float* __restrict__ cwt0,
    __bf16* __restrict__ w1f1, __bf16* __restrict__ w2f1, float* __restrict__ cwt1)
{
  int tid = threadIdx.x;
  if (blockIdx.x == 0) {
    for (int e = tid; e < 4 * 512; e += 256) {            // kb_w1 (14x64, K padded to 32)
      int t = e >> 9, L = (e >> 4) & 31, j = e & 15;
      int n = L & 15, kb0 = (L < 16) ? 0 : 8;
      int kk = kb0 + j + (j & 8);
      kbw1f[e] = (__bf16)((kk < POLY) ? kb_w1[kk * HID + t * 16 + n] : 0.0f);
    }
    for (int e = tid; e < 4 * 512; e += 256) {            // kb_w2 (64x32): t = kc*2+ct
      int t = e >> 9, L = (e >> 4) & 31, j = e & 15;
      int n = L & 15, kb0 = (L < 16) ? 0 : 8;
      int kc = t >> 1, ct = t & 1;
      kbw2f[e] = (__bf16)kb_w2[(kc * 32 + kb0 + j + (j & 8)) * BAS + ct * 16 + n];
    }
    for (int l = 0; l < 2; ++l) {
      const float* w1 = l ? w1_1 : w1_0;
      const float* w2 = l ? w2_1 : w2_0;
      const float* cw = l ? cw_1 : cw_0;
      __bf16* w1f = l ? w1f1 : w1f0;
      __bf16* w2f = l ? w2f1 : w2f0;
      float*  cwt = l ? cwt1 : cwt0;
      for (int e = tid; e < 32 * 512; e += 256) {         // w1 (64x256): t = hc*2+kc
        int t = e >> 9, L = (e >> 4) & 31, j = e & 15;
        int n = L & 15, kb0 = (L < 16) ? 0 : 8;
        int hc = t >> 1, kc = t & 1;
        w1f[e] = (__bf16)w1[(kc * 32 + kb0 + j + (j & 8)) * WHID + hc * 16 + n];
      }
      for (int e = tid; e < 32 * 512; e += 256) {         // w2 (256x64): t = kc*4+co
        int t = e >> 9, L = (e >> 4) & 31, j = e & 15;
        int n = L & 15, kb0 = (L < 16) ? 0 : 8;
        int kc = t >> 2, co = t & 3;
        w2f[e] = (__bf16)w2[(kc * 32 + kb0 + j + (j & 8)) * HID + co * 16 + n];
      }
      for (int e = tid; e < HID * BAS; e += 256) {        // conv_w transposed [c][k] fp32
        int c = e >> 5, k = e & 31;
        cwt[e] = cw[k * HID + c];
      }
    }
  }
  int total = BB * HID * NN;                              // stem, transposed output
  for (int idx = blockIdx.x * 256 + tid; idx < total; idx += gridDim.x * 256) {
    int b = idx / (HID * NN);
    int rem = idx - b * HID * NN;
    int c = rem / NN, s = rem - c * NN;
    float acc = 0.0f;
    const float* ap = a + (b * NN + s) * NIN;
#pragma unroll
    for (int i = 0; i < NIN; ++i) acc += ap[i] * stem_w[i * HID + c];
    ht0[idx] = (__bf16)acc;
  }
}

// ---------------------------------------------------------------------------
// kb: per 16-edge tile (fixed b,r; s0..s0+15): feats(14,padK32) -WMMA-> 64,
// gelu, LDS transpose, -WMMA-> 32, gelu, store kbt[b][r][k][s] bf16.
// ---------------------------------------------------------------------------
__global__ void __launch_bounds__(256) kb_kernel(
    const float* __restrict__ p, const float* __restrict__ kb_b1, const float* __restrict__ kb_b2,
    const __bf16* __restrict__ kbw1f, const __bf16* __restrict__ kbw2f,
    __bf16* __restrict__ kbt)
{
  __shared__ __bf16 lds[8][16][64];
  int tid = threadIdx.x;
  int wav = tid >> 5, lane = tid & 31;
  int hlf = lane >> 4, n = lane & 15, m = n;
  int koff = hlf * 8;

  Frag W1[4], W2[4];
#pragma unroll
  for (int t = 0; t < 4; ++t) {
    const __bf16* s1 = kbw1f + (t * 32 + lane) * 16;
    W1[t].q[0] = *(const uint4*)s1;  W1[t].q[1] = *(const uint4*)(s1 + 8);
    const __bf16* s2 = kbw2f + (t * 32 + lane) * 16;
    W2[t].q[0] = *(const uint4*)s2;  W2[t].q[1] = *(const uint4*)(s2 + 8);
  }
  float bias1[4], bias2[2];
#pragma unroll
  for (int t = 0; t < 4; ++t) bias1[t] = kb_b1[t * 16 + n];
#pragma unroll
  for (int t = 0; t < 2; ++t) bias2[t] = kb_b2[t * 16 + n];

  const int TT = BB * NN * (NN / 16);
  int stride = gridDim.x * 8;
  for (int t = blockIdx.x * 8 + wav; t < TT; t += stride) {
    int b = t / (NN * (NN / 16));
    int rem = t - b * NN * (NN / 16);
    int r = rem / (NN / 16);
    int s0 = (rem - r * (NN / 16)) * 16;

    int s = s0 + m;
    float x0 = p[(b * NN + r) * 2 + 0] - p[(b * NN + s) * 2 + 0];
    float x1 = p[(b * NN + r) * 2 + 1] - p[(b * NN + s) * 2 + 1];
    float f[14];
    f[0] = x0; f[1] = x1;
    f[2] = x0 * x0; f[3] = x0 * x1; f[4] = x1 * x0; f[5] = x1 * x1;
    f[6] = f[2] * x0; f[7] = f[2] * x1; f[8]  = f[3] * x0; f[9]  = f[3] * x1;
    f[10] = f[4] * x0; f[11] = f[4] * x1; f[12] = f[5] * x0; f[13] = f[5] * x1;

    // A-frag (bf16 16x32): lane<16 holds K{0..7,16..23}, lane>=16 K{8..15,24..31};
    // K>=14 are zero pad.
    Frag A;
    float g[8];
    g[0] = hlf ? f[8]  : f[0];  g[1] = hlf ? f[9]  : f[1];
    g[2] = hlf ? f[10] : f[2];  g[3] = hlf ? f[11] : f[3];
    g[4] = hlf ? f[12] : f[4];  g[5] = hlf ? f[13] : f[5];
    g[6] = hlf ? 0.0f  : f[6];  g[7] = hlf ? 0.0f  : f[7];
#pragma unroll
    for (int j = 0; j < 8; ++j) { A.h[j] = (__bf16)g[j]; A.h[8 + j] = (__bf16)0.0f; }

    f32x8 acc1[4];
#pragma unroll
    for (int t4 = 0; t4 < 4; ++t4) acc1[t4] = wmma_bf16(A.v, W1[t4].v, zero8());

#pragma unroll
    for (int t4 = 0; t4 < 4; ++t4)
#pragma unroll
      for (int v = 0; v < 8; ++v)
        lds[wav][v + 8 * hlf][t4 * 16 + n] = (__bf16)gelu_t(acc1[t4][v] + bias1[t4]);
    WAIT_DS();

    Frag A2[2];
#pragma unroll
    for (int kc = 0; kc < 2; ++kc) {
      A2[kc].q[0] = *(const uint4*)&lds[wav][m][kc * 32 + koff];
      A2[kc].q[1] = *(const uint4*)&lds[wav][m][kc * 32 + 16 + koff];
    }
    f32x8 acc2[2];
#pragma unroll
    for (int ct = 0; ct < 2; ++ct) {
      acc2[ct] = wmma_bf16(A2[0].v, W2[ct].v, zero8());
      acc2[ct] = wmma_bf16(A2[1].v, W2[2 + ct].v, acc2[ct]);
    }
#pragma unroll
    for (int ct = 0; ct < 2; ++ct) {
      union { uint4 q; __bf16 h[8]; } P;
#pragma unroll
      for (int v = 0; v < 8; ++v) P.h[v] = (__bf16)gelu_t(acc2[ct][v] + bias2[ct]);
      int k = ct * 16 + n;
      *(uint4*)&kbt[((size_t)(b * NN + r) * BAS + k) * NN + s0 + 8 * hlf] = P.q;
    }
  }
}

// ---------------------------------------------------------------------------
// conv: one wave per (b,r). G[k,c] = sum_s kb[b,r,s,k]*h[b,s,c] via 24x8 WMMAs
// (K=768), then h_pre[c] = sum_k conv_w_t[c][k]*G[k,c] + conv_b[c].
// ---------------------------------------------------------------------------
__global__ void __launch_bounds__(256) conv_kernel(
    const __bf16* __restrict__ kbt, const __bf16* __restrict__ ht_in,
    const float* __restrict__ cwt, const float* __restrict__ conv_b,
    float* __restrict__ hpre)
{
  int tid = threadIdx.x;
  int wav = tid >> 5, lane = tid & 31;
  int gw = blockIdx.x * 8 + wav;                 // exactly B*N waves
  int b = gw / NN, r = gw - b * NN;
  int hlf = lane >> 4, n = lane & 15, m = n;
  int koff = hlf * 8, kbase = hlf * 8;

  f32x8 acc[2][4];
#pragma unroll
  for (int kt = 0; kt < 2; ++kt)
#pragma unroll
    for (int ct = 0; ct < 4; ++ct) acc[kt][ct] = zero8();

  const __bf16* kbp = kbt + (size_t)(b * NN + r) * BAS * NN;
  const __bf16* hbp = ht_in + (size_t)b * HID * NN;

  for (int s0 = 0; s0 < NN; s0 += 32) {
    Frag Af[2], Bf[4];
#pragma unroll
    for (int kt = 0; kt < 2; ++kt) {
      const __bf16* src = kbp + (kt * 16 + m) * NN + s0;
      Af[kt].q[0] = *(const uint4*)(src + koff);
      Af[kt].q[1] = *(const uint4*)(src + 16 + koff);
      if (s0 + 32 < NN) __builtin_prefetch(src + 32, 0, 3);
    }
#pragma unroll
    for (int ct = 0; ct < 4; ++ct) {
      const __bf16* src = hbp + (ct * 16 + n) * NN + s0 + kbase;
      Bf[ct].q[0] = *(const uint4*)src;
      Bf[ct].q[1] = *(const uint4*)(src + 16);
    }
#pragma unroll
    for (int kt = 0; kt < 2; ++kt)
#pragma unroll
      for (int ct = 0; ct < 4; ++ct)
        acc[kt][ct] = wmma_bf16(Af[kt].v, Bf[ct].v, acc[kt][ct]);
  }

#pragma unroll
  for (int ct = 0; ct < 4; ++ct) {
    int c = ct * 16 + n;
    float part = 0.0f;
#pragma unroll
    for (int kt = 0; kt < 2; ++kt) {
      const float* cw = cwt + c * BAS + kt * 16 + 8 * hlf;
      float4 c0 = *(const float4*)cw;
      float4 c1 = *(const float4*)(cw + 4);
      part += c0.x * acc[kt][ct][0] + c0.y * acc[kt][ct][1] + c0.z * acc[kt][ct][2] + c0.w * acc[kt][ct][3]
            + c1.x * acc[kt][ct][4] + c1.y * acc[kt][ct][5] + c1.z * acc[kt][ct][6] + c1.w * acc[kt][ct][7];
    }
    part += __shfl_xor(part, 16, 32);            // combine k{0..7,16..23} with k{8..15,24..31}
    if (hlf == 0) hpre[(b * NN + r) * HID + c] = part + conv_b[c];
  }
}

// ---------------------------------------------------------------------------
// mlp: one wave per 16-row tile. LayerNorm -> (y@w1+b1) gelu -> @w2+b2,
// chained WMMAs with LDS transposes; writes next h transposed bf16 [b][c][r].
// ---------------------------------------------------------------------------
__global__ void __launch_bounds__(256) mlp_kernel(
    const float* __restrict__ hpre, const float* __restrict__ ln_s, const float* __restrict__ ln_b,
    const __bf16* __restrict__ w1f, const float* __restrict__ b1,
    const __bf16* __restrict__ w2f, const float* __restrict__ b2,
    __bf16* __restrict__ ht_out)
{
  __shared__ __bf16 yln[8][16][64];
  __shared__ __bf16 hbuf[8][16][32];
  int tid = threadIdx.x;
  int wav = tid >> 5, lane = tid & 31;
  int tile = blockIdx.x * 8 + wav;               // exactly B*N/16 tiles
  int b = tile / (NN / 16);
  int r0 = (tile - b * (NN / 16)) * 16;
  int hlf = lane >> 4, n = lane & 15, m = n;
  int koff = hlf * 8;

  // load 32 channels of one row per lane + LayerNorm stats via partner lane
  int r = r0 + m;
  const float* hp = hpre + (size_t)(b * NN + r) * HID + hlf * 32;
  float x[32];
#pragma unroll
  for (int j4 = 0; j4 < 8; ++j4) {
    float4 t = *(const float4*)(hp + j4 * 4);
    x[j4 * 4 + 0] = t.x; x[j4 * 4 + 1] = t.y; x[j4 * 4 + 2] = t.z; x[j4 * 4 + 3] = t.w;
  }
  float s1 = 0.0f, s2 = 0.0f;
#pragma unroll
  for (int j = 0; j < 32; ++j) { s1 += x[j]; s2 += x[j] * x[j]; }
  s1 += __shfl_xor(s1, 16, 32);
  s2 += __shfl_xor(s2, 16, 32);
  float mean = s1 * (1.0f / 64.0f);
  float var  = s2 * (1.0f / 64.0f) - mean * mean;
  float rstd = rsqrtf(var + 1e-6f);
#pragma unroll
  for (int j8 = 0; j8 < 4; ++j8) {
    union { uint4 q; __bf16 h[8]; } P;
#pragma unroll
    for (int j = 0; j < 8; ++j) {
      int c = hlf * 32 + j8 * 8 + j;
      P.h[j] = (__bf16)((x[j8 * 8 + j] - mean) * rstd * ln_s[c] + ln_b[c]);
    }
    *(uint4*)&yln[wav][m][hlf * 32 + j8 * 8] = P.q;
  }
  WAIT_DS();

  Frag A1[2];
#pragma unroll
  for (int kc = 0; kc < 2; ++kc) {
    A1[kc].q[0] = *(const uint4*)&yln[wav][m][kc * 32 + koff];
    A1[kc].q[1] = *(const uint4*)&yln[wav][m][kc * 32 + 16 + koff];
  }
  f32x8 oacc[4];
#pragma unroll
  for (int co = 0; co < 4; ++co) oacc[co] = zero8();

  for (int hp8 = 0; hp8 < 8; ++hp8) {            // hidden in pairs of 16-col tiles
#pragma unroll
    for (int u = 0; u < 2; ++u) {
      int hc = hp8 * 2 + u;
      Frag Wt0, Wt1;
      const __bf16* sp = w1f + ((hc * 2 + 0) * 32 + lane) * 16;
      Wt0.q[0] = *(const uint4*)sp;  Wt0.q[1] = *(const uint4*)(sp + 8);
      sp = w1f + ((hc * 2 + 1) * 32 + lane) * 16;
      Wt1.q[0] = *(const uint4*)sp;  Wt1.q[1] = *(const uint4*)(sp + 8);
      f32x8 hacc = wmma_bf16(A1[0].v, Wt0.v, zero8());
      hacc = wmma_bf16(A1[1].v, Wt1.v, hacc);
      float bias = b1[hc * 16 + n];
#pragma unroll
      for (int v = 0; v < 8; ++v)
        hbuf[wav][v + 8 * hlf][u * 16 + n] = (__bf16)gelu_t(hacc[v] + bias);
    }
    WAIT_DS();
    Frag A2;
    A2.q[0] = *(const uint4*)&hbuf[wav][m][koff];
    A2.q[1] = *(const uint4*)&hbuf[wav][m][16 + koff];
#pragma unroll
    for (int co = 0; co < 4; ++co) {
      Frag Wt;
      const __bf16* sp = w2f + ((hp8 * 4 + co) * 32 + lane) * 16;
      Wt.q[0] = *(const uint4*)sp;  Wt.q[1] = *(const uint4*)(sp + 8);
      oacc[co] = wmma_bf16(A2.v, Wt.v, oacc[co]);
    }
  }

#pragma unroll
  for (int co = 0; co < 4; ++co) {
    int c = co * 16 + n;
    float bias = b2[c];
    union { uint4 q; __bf16 h[8]; } Q;
#pragma unroll
    for (int v = 0; v < 8; ++v) Q.h[v] = (__bf16)(oacc[co][v] + bias);
    *(uint4*)&ht_out[((size_t)b * HID + c) * NN + r0 + 8 * hlf] = Q.q;
  }
}

// ---------------------------------------------------------------------------
// readout: out[b] = mean_r sum_c h[b,r,c]*ro_w[c]
// ---------------------------------------------------------------------------
__global__ void __launch_bounds__(256) readout_kernel(
    const __bf16* __restrict__ ht, const float* __restrict__ ro_w, float* __restrict__ out)
{
  __shared__ float red[256];
  int b = blockIdx.x, tid = threadIdx.x;
  float acc = 0.0f;
  const __bf16* base = ht + (size_t)b * HID * NN;
  for (int idx = tid; idx < HID * NN; idx += 256)
    acc += (float)base[idx] * ro_w[idx / NN];
  red[tid] = acc;
  __syncthreads();
  for (int s = 128; s > 0; s >>= 1) {
    if (tid < s) red[tid] += red[tid + s];
    __syncthreads();
  }
  if (tid == 0) out[b] = red[0] * (1.0f / NN);
}

// ---------------------------------------------------------------------------
extern "C" void kernel_launch(void* const* d_in, const int* in_sizes, int n_in,
                              void* d_out, int out_size, void* d_ws, size_t ws_size,
                              hipStream_t stream) {
  const float* p      = (const float*)d_in[0];
  const float* a      = (const float*)d_in[1];
  const float* kb_w1  = (const float*)d_in[2];
  const float* kb_b1  = (const float*)d_in[3];
  const float* kb_w2  = (const float*)d_in[4];
  const float* kb_b2  = (const float*)d_in[5];
  const float* stem_w = (const float*)d_in[6];
  const float* cw0  = (const float*)d_in[7];
  const float* cb0  = (const float*)d_in[8];
  const float* lns0 = (const float*)d_in[9];
  const float* lnb0 = (const float*)d_in[10];
  const float* w1_0 = (const float*)d_in[11];
  const float* b1_0 = (const float*)d_in[12];
  const float* w2_0 = (const float*)d_in[13];
  const float* b2_0 = (const float*)d_in[14];
  const float* cw1  = (const float*)d_in[15];
  const float* cb1  = (const float*)d_in[16];
  const float* lns1 = (const float*)d_in[17];
  const float* lnb1 = (const float*)d_in[18];
  const float* w1_1 = (const float*)d_in[19];
  const float* b1_1 = (const float*)d_in[20];
  const float* w2_1 = (const float*)d_in[21];
  const float* b2_1 = (const float*)d_in[22];
  const float* ro_w = (const float*)d_in[23];
  float* out = (float*)d_out;

  char* ws = (char*)d_ws;
  __bf16* kbt   = (__bf16*)(ws);                 // 75,497,472 B: kb transposed [b][r][k][s] bf16
  __bf16* hta   = (__bf16*)(ws + 75497472);      // 196,608 B: h transposed [b][c][s] bf16
  __bf16* htb   = (__bf16*)(ws + 75694080);      // 196,608 B
  float*  hpre  = (float*) (ws + 75890688);      // 393,216 B
  __bf16* kbw1f = (__bf16*)(ws + 76283904);      // 4,096 B
  __bf16* kbw2f = (__bf16*)(ws + 76288000);      // 4,096 B
  __bf16* w1f0  = (__bf16*)(ws + 76292096);      // 32,768 B
  __bf16* w2f0  = (__bf16*)(ws + 76324864);      // 32,768 B
  __bf16* w1f1  = (__bf16*)(ws + 76357632);      // 32,768 B
  __bf16* w2f1  = (__bf16*)(ws + 76390400);      // 32,768 B
  float*  cwt0  = (float*) (ws + 76423168);      // 8,192 B
  float*  cwt1  = (float*) (ws + 76431360);      // 8,192 B -> total 76,439,552 B
  if (ws_size < (size_t)76439552) return;

  prep_kernel<<<64, 256, 0, stream>>>(a, stem_w, kb_w1, kb_w2, w1_0, w2_0, cw0,
                                      w1_1, w2_1, cw1, hta, kbw1f, kbw2f,
                                      w1f0, w2f0, cwt0, w1f1, w2f1, cwt1);
  kb_kernel<<<2304, 256, 0, stream>>>(p, kb_b1, kb_b2, kbw1f, kbw2f, kbt);
  conv_kernel<<<192, 256, 0, stream>>>(kbt, hta, cwt0, cb0, hpre);
  mlp_kernel<<<12, 256, 0, stream>>>(hpre, lns0, lnb0, w1f0, b1_0, w2f0, b2_0, htb);
  conv_kernel<<<192, 256, 0, stream>>>(kbt, htb, cwt1, cb1, hpre);
  mlp_kernel<<<12, 256, 0, stream>>>(hpre, lns1, lnb1, w1f1, b1_1, w2f1, b2_1, hta);
  readout_kernel<<<2, 256, 0, stream>>>(hta, ro_w, out);
}
